// EncoderRNN_72009421684946
// MI455X (gfx1250) — matrix operand order repaired and checked by reading, not compiled, not run on hardware
//
#include <hip/hip_runtime.h>
#include <hip/hip_bf16.h>
#include <math.h>

// Problem constants (match reference)
#define BB   32
#define TT   1000
#define DD   240
#define HH   256
#define LLAY 3
#define G3H  768            // 3*H
#define NTOK (BB*TT)        // 32000 tokens
#define EPSV 1e-5f

typedef __bf16 bf16;
typedef __attribute__((ext_vector_type(16))) __bf16 v16bf;
typedef __attribute__((ext_vector_type(8)))  __bf16 v8bf;
typedef __attribute__((ext_vector_type(8)))  float  v8f;

union Frag {
  v16bf v;
  v8bf  h[2];
};

__device__ __forceinline__ v8f wmma_bf16(const Frag& a, const Frag& b, v8f c) {
  // D = A(16x32 bf16) x B(32x16 bf16) + C(16x16 f32)
  return __builtin_amdgcn_wmma_f32_16x16x32_bf16(false, a.v, false, b.v,
                                                 (short)0, c, false, false);
}

// gfx1250 async global->LDS copy (ASYNCcnt path). 16 bytes per lane.
__device__ __forceinline__ void async_copy_b128(void* lds_dst, const void* gsrc) {
  unsigned lds = (unsigned)(unsigned long long)lds_dst;
  unsigned long long ga = (unsigned long long)gsrc;
  asm volatile("global_load_async_to_lds_b128 %0, %1, off"
               :: "v"(lds), "v"(ga) : "memory");
}

__device__ __forceinline__ float fast_sigmoid(float x) {
  return __builtin_amdgcn_rcpf(1.0f + __expf(-x));
}
__device__ __forceinline__ float fast_tanh(float x) {
  // tanh(x) = 1 - 2/(exp(2x)+1); correct limits at +/-inf.
  return 1.0f - 2.0f * __builtin_amdgcn_rcpf(__expf(2.0f * x) + 1.0f);
}

// ---------------------------------------------------------------------------
// fp32 -> bf16 cast
// ---------------------------------------------------------------------------
__global__ void cast_f32_bf16(const float* __restrict__ src,
                              bf16* __restrict__ dst, long n) {
  long i = (long)blockIdx.x * blockDim.x + threadIdx.x;
  if (i < n) dst[i] = (bf16)src[i];
}

// ---------------------------------------------------------------------------
// Bulk GEMM:  C[M,N] = A[M,K](bf16) @ W[N,K]^T(bf16) + bias[N]
// Block tile 64M x 128N, 8 waves, each wave 32x32 (2x2 WMMA tiles).
// Double-buffered LDS staged with global_load_async_to_lds_b128; in-order
// async retirement means waiting down to the next tile's in-flight count
// guarantees the current tile is resident.
// KTAIL=false: K % 32 == 0, branch-free staging.
// KTAIL=true:  K % 8 == 0, tail chunks zero-filled in LDS.
// GATE_ILV=true: output column (gate-major, 3*256) is stored interleaved as
// unit*3+gate so the GRU scan reads r/z/n with one b96 load.
// ---------------------------------------------------------------------------
#define ASTR 40
#define BSTR 40
template <bool KTAIL, bool GATE_ILV>
__global__ __launch_bounds__(256)
void gemm_bias_wmma(const bf16* __restrict__ A, int lda,
                    const bf16* __restrict__ W, int ldw,
                    const float* __restrict__ bias,
                    float* __restrict__ C, int ldc, int K) {
  __shared__ bf16 As[2][64 * ASTR];
  __shared__ bf16 Bs[2][128 * BSTR];

  const int tid    = threadIdx.x;
  const int lane   = tid & 31;
  const int wv     = tid >> 5;
  const int lane15 = lane & 15;
  const int khalf  = lane >> 4;
  const int wm     = wv >> 2;   // 0..1 (M sub-block)
  const int wn     = wv & 3;    // 0..3 (N sub-block)
  const long blockM = (long)blockIdx.x * 64;
  const long blockN = (long)blockIdx.y * 128;

  const int srow = tid >> 2;    // staging row (0..63)
  const int sch  = tid & 3;     // staging 16B chunk (0..3)

  // Issue async copies for K-step `ks` into buffer `buf`.
  // Returns number of async loads issued by this thread (3, or 0 on tail).
  auto stage = [&](int buf, int ks) -> int {
    const int kk = ks * 32 + sch * 8;
    if (KTAIL && kk + 8 > K) {
      v8bf z = {};
      *(v8bf*)&As[buf][srow * ASTR + sch * 8] = z;
      *(v8bf*)&Bs[buf][srow * BSTR + sch * 8] = z;
      *(v8bf*)&Bs[buf][(srow + 64) * BSTR + sch * 8] = z;
      return 0;
    }
    async_copy_b128(&As[buf][srow * ASTR + sch * 8],
                    A + (blockM + srow) * (long)lda + kk);
    async_copy_b128(&Bs[buf][srow * BSTR + sch * 8],
                    W + (blockN + srow) * (long)ldw + kk);
    async_copy_b128(&Bs[buf][(srow + 64) * BSTR + sch * 8],
                    W + (blockN + srow + 64) * (long)ldw + kk);
    return 3;
  };

  v8f acc[2][2];
  acc[0][0] = {}; acc[0][1] = {}; acc[1][0] = {}; acc[1][1] = {};

  const int ksteps = (K + 31) / 32;
  stage(0, 0);
  for (int ks = 0; ks < ksteps; ++ks) {
    const int cur = ks & 1;
    int nxt = 0;
    if (ks + 1 < ksteps) nxt = stage(cur ^ 1, ks + 1);
    if (nxt) asm volatile("s_wait_asynccnt 0x3" ::: "memory");
    else     asm volatile("s_wait_asynccnt 0x0" ::: "memory");
    __syncthreads();

    Frag a[2], b[2];
    #pragma unroll
    for (int mt = 0; mt < 2; ++mt) {
      const bf16* p = &As[cur][(wm * 32 + mt * 16 + lane15) * ASTR + khalf * 16];
      a[mt].h[0] = *(const v8bf*)p;         // K 0..7  of lane's half
      a[mt].h[1] = *(const v8bf*)(p + 8);   // K 8..15
    }
    #pragma unroll
    for (int nt = 0; nt < 2; ++nt) {
      const bf16* p = &Bs[cur][(wn * 32 + nt * 16 + lane15) * BSTR + khalf * 16];
      b[nt].h[0] = *(const v8bf*)p;         // K 0..7  of lane's half
      b[nt].h[1] = *(const v8bf*)(p + 8);   // K 8..15
    }
    #pragma unroll
    for (int mt = 0; mt < 2; ++mt)
      #pragma unroll
      for (int nt = 0; nt < 2; ++nt)
        acc[mt][nt] = wmma_bf16(a[mt], b[nt], acc[mt][nt]);
    __syncthreads();   // all waves done with buf `cur` before it is re-staged
  }

  // epilogue: D layout -> VGPR v holds row m = v + 8*khalf, col = lane15
  #pragma unroll
  for (int mt = 0; mt < 2; ++mt) {
    #pragma unroll
    for (int nt = 0; nt < 2; ++nt) {
      long ncol = blockN + wn * 32 + nt * 16 + lane15;
      float bv = bias ? bias[ncol] : 0.0f;
      long coff;
      if (GATE_ILV) {
        long unit = ncol & (HH - 1);
        long gate = ncol >> 8;
        coff = unit * 3 + gate;           // [unit][gate] interleave
      } else {
        coff = ncol;
      }
      #pragma unroll
      for (int v = 0; v < 8; ++v) {
        long mrow = blockM + wm * 32 + mt * 16 + v + 8 * khalf;
        C[mrow * (long)ldc + coff] = acc[mt][nt][v] + bv;
      }
    }
  }
}

// ---------------------------------------------------------------------------
// Masked BatchNorm statistics: one block per channel.
// ---------------------------------------------------------------------------
__global__ __launch_bounds__(256)
void bn_stats(const float* __restrict__ x, int width,
              const int* __restrict__ xlens,
              float* __restrict__ mean, float* __restrict__ rstd) {
  const int c = blockIdx.x;
  __shared__ float s1s[256], s2s[256];
  __shared__ int   lens_s[BB];
  const int tid = threadIdx.x;
  if (tid < BB) lens_s[tid] = xlens[tid];
  __syncthreads();

  float s1 = 0.f, s2 = 0.f;
  for (int i = tid; i < NTOK; i += 256) {
    int b = i / TT, t = i - b * TT;
    if (t < lens_s[b]) {
      float v = x[(long)i * width + c];
      s1 += v; s2 += v * v;
    }
  }
  s1s[tid] = s1; s2s[tid] = s2;
  __syncthreads();
  for (int off = 128; off > 0; off >>= 1) {
    if (tid < off) { s1s[tid] += s1s[tid + off]; s2s[tid] += s2s[tid + off]; }
    __syncthreads();
  }
  if (tid == 0) {
    float n = 0.f;
    for (int b = 0; b < BB; ++b) n += (float)lens_s[b];
    float mu  = s1s[0] / n;
    float var = s2s[0] / n - mu * mu;
    mean[c] = mu;
    rstd[c] = rsqrtf(var + EPSV);
  }
}

// ---------------------------------------------------------------------------
// BN apply (masked) fused with cast to bf16 for the following GEMM.
// ---------------------------------------------------------------------------
__global__ void bn_apply_cast(const float* __restrict__ x, int width,
                              const int* __restrict__ xlens,
                              const float* __restrict__ mean,
                              const float* __restrict__ rstd,
                              const float* __restrict__ g,
                              const float* __restrict__ bcoef,
                              bf16* __restrict__ y) {
  long i = (long)blockIdx.x * blockDim.x + threadIdx.x;
  long total = (long)NTOK * width;
  if (i >= total) return;
  int  c   = (int)(i % width);
  long row = i / width;
  int  b   = (int)(row / TT);
  int  t   = (int)(row - (long)b * TT);
  float v = 0.0f;
  if (t < xlens[b]) v = (x[i] - mean[c]) * rstd[c] * g[c] + bcoef[c];
  y[i] = (bf16)v;
}

// ---------------------------------------------------------------------------
// Sequential GRU scan. Grid = 4 blocks: {dir, batch half}. 512 threads.
// Per step: GEMM h(16x256) @ W_hh^T(256x768) via WMMA.
// W_hh^T split to avoid VGPR spills (previous round spilled to scratch):
//   - gates r,z: B-fragments resident in VGPRs (Bf[2][8] = 128 VGPRs)
//   - gate  n : B-fragments cached in LDS (128 KB, per-wave regions,
//               16B lane stride -> conflict-free ds_load_b128 pairs)
// h staged in LDS as bf16 A-tiles; each lane owns h(rows v+8*khalf, its
// column) in registers across all 1000 steps. gx is interleaved
// [token][unit][3] -> one b96 load per row, issued BEFORE the WMMA phase so
// L2 latency hides under matrix work. Backward dir reads/writes token
// j = len-1-s, so outputs land re-reversed.
// ---------------------------------------------------------------------------
#define HSTR 264   // padded stride (528B) -> conflict-free A-tile reads
__global__ __launch_bounds__(512)
void gru_scan(const float* __restrict__ gx_fw, const float* __restrict__ gx_bw,
              const bf16*  __restrict__ whh_fw, const bf16* __restrict__ whh_bw,
              const float* __restrict__ bhh_fw, const float* __restrict__ bhh_bw,
              const float* __restrict__ h0_fw,  const float* __restrict__ h0_bw,
              const int*   __restrict__ xlens,
              float* __restrict__ out) {
  __shared__ bf16 hsm[16 * HSTR];                 // 8.4 KB h tile
  __shared__ bf16 bns[16 * 8 * 2 * 32 * 8];       // 128 KB gate-n B cache

  const int dir   = blockIdx.x >> 1;
  const int mtile = blockIdx.x & 1;
  const int tid    = threadIdx.x;
  const int lane   = tid & 31;
  const int w      = tid >> 5;        // wave 0..15
  const int lane15 = lane & 15;
  const int khalf  = lane >> 4;

  const float* gx  = dir ? gx_bw  : gx_fw;
  const bf16*  whh = dir ? whh_bw : whh_fw;
  const float* bhh = dir ? bhh_bw : bhh_fw;
  const float* h0  = dir ? h0_bw  : h0_fw;

  const int col = w * 16 + lane15;    // hidden unit owned by this lane

  // Resident W_hh^T B-fragments for gates r,z (128 VGPRs)
  Frag Bf[2][8];
  #pragma unroll
  for (int g = 0; g < 2; ++g) {
    const int nrow = g * 256 + col;   // row of w_hh (output gate unit)
    #pragma unroll
    for (int ks = 0; ks < 8; ++ks) {
      const bf16* p = whh + (long)nrow * HH + ks * 32 + khalf * 16;
      Bf[g][ks].h[0] = *(const v8bf*)p;
      Bf[g][ks].h[1] = *(const v8bf*)(p + 8);
    }
  }
  // Gate-n B-fragments -> LDS; wave w owns region w. Slot layout:
  // elem index = (((w*8+ks)*2 + half)*32 + lane)*8   (16B per lane per half)
  #pragma unroll
  for (int ks = 0; ks < 8; ++ks) {
    const bf16* p = whh + (long)(512 + col) * HH + ks * 32 + khalf * 16;
    *(v8bf*)&bns[(((w * 8 + ks) * 2 + 0) * 32 + lane) * 8] = *(const v8bf*)p;
    *(v8bf*)&bns[(((w * 8 + ks) * 2 + 1) * 32 + lane) * 8] = *(const v8bf*)(p + 8);
  }
  const float br = bhh[col];
  const float bz = bhh[256 + col];
  const float bn = bhh[512 + col];

  // Per-lane hidden state (rows m = v + 8*khalf of this block's 16 batches)
  float hreg[8];
  int   lenr[8];
  #pragma unroll
  for (int v = 0; v < 8; ++v) {
    hreg[v] = h0[col];                               // broadcast init state
    lenr[v] = xlens[mtile * 16 + 8 * khalf + v];
  }
  #pragma unroll
  for (int v = 0; v < 8; ++v)
    hsm[(8 * khalf + v) * HSTR + col] = (bf16)hreg[v];
  __syncthreads();

  for (int s = 0; s < TT; ++s) {
    // --- issue gx gathers for this step first (hide L2 latency) ---
    int    jrow[8];
    float3 gv[8];
    #pragma unroll
    for (int v = 0; v < 8; ++v) {
      const int bm  = mtile * 16 + 8 * khalf + v;
      const int len = lenr[v];
      const int j   = dir ? ((s < len) ? (len - 1 - s) : s) : s;
      jrow[v] = j;
      gv[v] = *(const float3*)(gx + (((long)bm * TT + j) * HH + col) * 3);
    }

    // --- recurrent GEMM: gh = h @ W_hh^T ---
    v8f ar = {}, az = {}, an = {};
    #pragma unroll
    for (int ks = 0; ks < 8; ++ks) {
      Frag A;
      const bf16* p = hsm + lane15 * HSTR + ks * 32 + khalf * 16;
      A.h[0] = *(const v8bf*)p;         // K 0..7  of lane's half
      A.h[1] = *(const v8bf*)(p + 8);   // K 8..15
      Frag Bn;
      Bn.h[0] = *(const v8bf*)&bns[(((w * 8 + ks) * 2 + 0) * 32 + lane) * 8];
      Bn.h[1] = *(const v8bf*)&bns[(((w * 8 + ks) * 2 + 1) * 32 + lane) * 8];
      ar = wmma_bf16(A, Bf[0][ks], ar);
      az = wmma_bf16(A, Bf[1][ks], az);
      an = wmma_bf16(A, Bn, an);
    }
    __syncthreads();   // all waves done reading hsm

    // --- gate math + state update ---
    #pragma unroll
    for (int v = 0; v < 8; ++v) {
      const int m_loc = 8 * khalf + v;
      const int bm    = mtile * 16 + m_loc;
      const int len   = lenr[v];
      float r  = fast_sigmoid(gv[v].x + ar[v] + br);
      float z  = fast_sigmoid(gv[v].y + az[v] + bz);
      float nn = fast_tanh(gv[v].z + r * (an[v] + bn));
      float hnew  = (1.0f - z) * nn + z * hreg[v];
      bool  valid = (s < len);
      hreg[v] = valid ? hnew : hreg[v];
      out[((long)bm * TT + jrow[v]) * (2 * HH) + dir * HH + col] =
          valid ? hreg[v] : 0.0f;
      hsm[m_loc * HSTR + col] = (bf16)hreg[v];
    }
    __syncthreads();   // h ready for next step's A-tiles
  }
}

// ---------------------------------------------------------------------------
// Host launcher
// ---------------------------------------------------------------------------
extern "C" void kernel_launch(void* const* d_in, const int* in_sizes, int n_in,
                              void* d_out, int out_size, void* d_ws, size_t ws_size,
                              hipStream_t stream) {
  (void)in_sizes; (void)n_in; (void)out_size; (void)ws_size;

  const float* xs         = (const float*)d_in[0];
  const int*   xlens      = (const int*)  d_in[1];
  const float* embed_w    = (const float*)d_in[2];
  const float* embed_b    = (const float*)d_in[3];
  const float* init_state = (const float*)d_in[4];
  // per layer: bn_g, bn_b, fw{w_ih,w_hh,b_ih,b_hh}, bw{w_ih,w_hh,b_ih,b_hh}
  auto LP = [&](int l, int k) { return (const float*)d_in[5 + l * 10 + k]; };

  // ---- workspace carve-up ----
  size_t off = 0;
  auto carve = [&](size_t bytes) {
    void* p = (char*)d_ws + off;
    off += (bytes + 255) & ~(size_t)255;
    return p;
  };
  bf16* xs_bf = (bf16*)carve((size_t)NTOK * DD * 2);
  bf16* ew_bf = (bf16*)carve((size_t)HH * DD * 2);
  bf16* wih_bf[LLAY][2];
  bf16* whh_bf[LLAY][2];
  for (int l = 0; l < LLAY; ++l) {
    int insize = (l == 0) ? HH : 2 * HH;
    for (int d = 0; d < 2; ++d) {
      wih_bf[l][d] = (bf16*)carve((size_t)G3H * insize * 2);
      whh_bf[l][d] = (bf16*)carve((size_t)G3H * HH * 2);
    }
  }
  float* bufA  = (float*)carve((size_t)NTOK * 512 * 4);
  float* bufB  = (float*)carve((size_t)NTOK * 512 * 4);
  bf16*  xn_bf = (bf16*) carve((size_t)NTOK * 512 * 2);
  float* gx_f  = (float*)carve((size_t)NTOK * G3H * 4);
  float* gx_b  = (float*)carve((size_t)NTOK * G3H * 4);
  float* meanp = (float*)carve(512 * 4);
  float* rstdp = (float*)carve(512 * 4);

  auto cast = [&](const float* s, bf16* d, long n) {
    cast_f32_bf16<<<dim3((unsigned)((n + 255) / 256)), 256, 0, stream>>>(s, d, n);
  };

  // ---- weight / input casts ----
  cast(xs, xs_bf, (long)NTOK * DD);
  cast(embed_w, ew_bf, (long)HH * DD);
  for (int l = 0; l < LLAY; ++l) {
    int insize = (l == 0) ? HH : 2 * HH;
    cast(LP(l, 2), wih_bf[l][0], (long)G3H * insize);  // fw w_ih
    cast(LP(l, 3), whh_bf[l][0], (long)G3H * HH);      // fw w_hh
    cast(LP(l, 6), wih_bf[l][1], (long)G3H * insize);  // bw w_ih
    cast(LP(l, 7), whh_bf[l][1], (long)G3H * HH);      // bw w_hh
  }

  // ---- embedding: x0 = xs @ embed_w^T + embed_b  -> bufA [NTOK, 256] ----
  gemm_bias_wmma<true, false><<<dim3(NTOK / 64, HH / 128), 256, 0, stream>>>(
      xs_bf, DD, ew_bf, DD, embed_b, bufA, HH, DD);

  float* x  = bufA;
  float* xn = bufB;
  for (int l = 0; l < LLAY; ++l) {
    const int insize = (l == 0) ? HH : 2 * HH;

    // masked BN -> bf16 activations
    bn_stats<<<dim3(insize), 256, 0, stream>>>(x, insize, xlens, meanp, rstdp);
    long tot = (long)NTOK * insize;
    bn_apply_cast<<<dim3((unsigned)((tot + 255) / 256)), 256, 0, stream>>>(
        x, insize, xlens, meanp, rstdp, LP(l, 0), LP(l, 1), xn_bf);

    // gx = x_bf @ w_ih^T + b_ih, stored gate-interleaved for the scan
    gemm_bias_wmma<false, true><<<dim3(NTOK / 64, G3H / 128), 256, 0, stream>>>(
        xn_bf, insize, wih_bf[l][0], insize, LP(l, 4), gx_f, G3H, insize);
    gemm_bias_wmma<false, true><<<dim3(NTOK / 64, G3H / 128), 256, 0, stream>>>(
        xn_bf, insize, wih_bf[l][1], insize, LP(l, 8), gx_b, G3H, insize);

    // sequential bidirectional scan
    float* outp = (l == LLAY - 1) ? (float*)d_out : xn;
    gru_scan<<<dim3(4), 512, 0, stream>>>(
        gx_f, gx_b, whh_bf[l][0], whh_bf[l][1], LP(l, 5), LP(l, 9),
        init_state + (size_t)(2 * l) * HH, init_state + (size_t)(2 * l + 1) * HH,
        xlens, outp);

    float* t = x; x = xn; xn = t;
  }
}
